// NCAPerception2d_15023795601708
// MI455X (gfx1250) — matrix operand order repaired and checked by reading, not compiled, not run on hardware
//
#include <hip/hip_runtime.h>
#include <hip/hip_bf16.h>

typedef __attribute__((ext_vector_type(16))) __bf16 v16bf;
typedef __attribute__((ext_vector_type(8)))  float  v8f;

#define C_IN   64
#define F_DIM  128
#define HW     65536          // 256*256
#define TW     128            // pixels (w-direction) per block tile
#define XCOLS  130            // TW + 2 halo
#define XSTR   72             // padded channel stride in shorts (144B, 16B aligned)

__device__ __forceinline__ unsigned short f2bf(float f) {
    unsigned u = __float_as_uint(f);
    unsigned r = u + 0x7FFFu + ((u >> 16) & 1u);   // round-to-nearest-even
    return (unsigned short)(r >> 16);
}

// ---- prepack: W_learned (F,C,3,3) f32 -> Wb[t][f][c] bf16 -------------------
__global__ void prepack_weights(const float* __restrict__ Wl,
                                unsigned short* __restrict__ Wb) {
    int i = blockIdx.x * blockDim.x + threadIdx.x;
    if (i >= 9 * F_DIM * C_IN) return;
    int c  = i & (C_IN - 1);
    int tf = i >> 6;
    int f  = tf & (F_DIM - 1);
    int t  = tf >> 7;
    Wb[i] = f2bf(Wl[(f * C_IN + c) * 9 + t]);
}

// ---- depthwise fixed filters (identity / sobel_y / sobel_x / laplacian) -----
__global__ void depthwise_kernel(const float* __restrict__ x,
                                 float* __restrict__ out) {
    int idx = blockIdx.x * blockDim.x + threadIdx.x;   // N*C*H*W threads
    int w = idx & 255, h = (idx >> 8) & 255, c = (idx >> 16) & 63, n = idx >> 22;
    const float* xp = x + ((size_t)(n * C_IN + c) << 16);
    int hm = h ? h - 1 : 0, hp = h < 255 ? h + 1 : 255;
    int wm = w ? w - 1 : 0, wp = w < 255 ? w + 1 : 255;
    float a00 = xp[hm * 256 + wm], a01 = xp[hm * 256 + w], a02 = xp[hm * 256 + wp];
    float a10 = xp[h  * 256 + wm], a11 = xp[h  * 256 + w], a12 = xp[h  * 256 + wp];
    float a20 = xp[hp * 256 + wm], a21 = xp[hp * 256 + w], a22 = xp[hp * 256 + wp];
    float sy  = (a00 + 2.f * a01 + a02) - (a20 + 2.f * a21 + a22);
    float sx  = (a00 - a02) + 2.f * (a10 - a12) + (a20 - a22);
    float lap = 4.f * a11 - a01 - a10 - a12 - a21;
    size_t hw = (size_t)h * 256 + w;
    size_t nb = (size_t)n * 384 * HW;
    out[nb + (size_t)(c * 3 + 0) * HW + hw] = a11;
    out[nb + (size_t)(c * 3 + 1) * HW + hw] = sy;
    out[nb + (size_t)(c * 3 + 2) * HW + hw] = sx;
    out[nb + (size_t)(192 + c)   * HW + hw] = lap;
}

// ---- learned conv: implicit GEMM via v_wmma_f32_16x16x32_bf16 ---------------
// Block: 256 threads = 8 waves. Tile: 128 pixels x 128 features.
// Wave (wm,wn): wm in {0,1} -> 64-pixel group, wn in {0..3} -> 32-feature group.
__global__ __launch_bounds__(256)
void learned_conv_wmma(const float* __restrict__ x,
                       const unsigned short* __restrict__ Wb,
                       const float* __restrict__ bias,
                       float* __restrict__ out) {
    __shared__ __align__(16) unsigned short Xs[3 * XCOLS * XSTR];

    int tid = threadIdx.x;
    int w0  = blockIdx.x * TW;
    int h   = blockIdx.y;
    int n   = blockIdx.z;

    // Stage halo tile (3 rows x 130 cols x 64 ch) as bf16, channel-contiguous.
    const float* xb = x + ((size_t)n * C_IN << 16);
    for (int i = tid; i < C_IN * 3 * XCOLS; i += 256) {
        int c   = i / (3 * XCOLS);
        int s   = i - c * 3 * XCOLS;
        int r   = s / XCOLS;
        int col = s - r * XCOLS;
        int hh = h - 1 + r;  hh = hh < 0 ? 0 : (hh > 255 ? 255 : hh);
        int ww = w0 - 1 + col; ww = ww < 0 ? 0 : (ww > 255 ? 255 : ww);
        Xs[(r * XCOLS + col) * XSTR + c] = f2bf(xb[((size_t)c << 16) + hh * 256 + ww]);
    }
    __syncthreads();

    int lane = tid & 31;
    int wave = tid >> 5;
    int wm   = wave & 1;
    int wn   = wave >> 1;
    int half = lane >> 4;
    int l15  = lane & 15;

    // Seed accumulators with bias (D column N == lane&15 == feature).
    v8f acc[4][2];
    for (int ns = 0; ns < 2; ++ns) {
        float bv = bias[wn * 32 + ns * 16 + l15];
        for (int ms = 0; ms < 4; ++ms)
            for (int j = 0; j < 8; ++j) acc[ms][ns][j] = bv;
    }

    for (int t = 0; t < 9; ++t) {
        int dh = t / 3 - 1, dw = t % 3 - 1;
        int r  = 1 + dh;
        if (t < 8) __builtin_prefetch(Wb + (t + 1) * F_DIM * C_IN, 0, 3);
        for (int kb = 0; kb < 2; ++kb) {
            // B fragments: K = channels contiguous per feature (L2-resident).
            union { v16bf v; uint4 q[2]; } bfr[2];
            for (int ns = 0; ns < 2; ++ns) {
                int f = wn * 32 + ns * 16 + l15;
                const unsigned short* bp =
                    Wb + ((t * F_DIM + f) * C_IN + kb * 32 + half * 8);
                bfr[ns].q[0] = *(const uint4*)(bp);
                bfr[ns].q[1] = *(const uint4*)(bp + 16);
            }
            // A fragments: lane-half 0 -> K +0..7 / +16..23, half 1 -> +8..15 / +24..31.
            union { v16bf v; uint4 q[2]; } afr[4];
            for (int ms = 0; ms < 4; ++ms) {
                int px  = wm * 64 + ms * 16 + l15;
                int col = 1 + px + dw;
                const unsigned short* ap =
                    &Xs[(r * XCOLS + col) * XSTR + kb * 32 + half * 8];
                afr[ms].q[0] = *(const uint4*)(ap);
                afr[ms].q[1] = *(const uint4*)(ap + 16);
            }
            for (int ms = 0; ms < 4; ++ms)
                for (int ns = 0; ns < 2; ++ns)
                    acc[ms][ns] = __builtin_amdgcn_wmma_f32_16x16x32_bf16(
                        false, afr[ms].v, false, bfr[ns].v,
                        (short)0, acc[ms][ns], false, false);
        }
    }

    // Store D into the "learned" slab: out channel = 256 + feature.
    size_t nb = (size_t)n * 384 * HW;
    for (int ns = 0; ns < 2; ++ns) {
        int feat = wn * 32 + ns * 16 + l15;
        float* op = out + nb + (size_t)(256 + feat) * HW + (size_t)h * 256;
        for (int ms = 0; ms < 4; ++ms) {
            int wbase = w0 + wm * 64 + ms * 16 + half * 8;
            for (int j = 0; j < 8; ++j) op[wbase + j] = acc[ms][ns][j];
        }
    }
}

extern "C" void kernel_launch(void* const* d_in, const int* in_sizes, int n_in,
                              void* d_out, int out_size, void* d_ws, size_t ws_size,
                              hipStream_t stream) {
    const float* x    = (const float*)d_in[0];   // (8,64,256,256)
    const float* Wl   = (const float*)d_in[1];   // (128,64,3,3)
    const float* bias = (const float*)d_in[2];   // (128,)
    float* out = (float*)d_out;                  // (8,384,256,256)
    unsigned short* Wb = (unsigned short*)d_ws;  // 9*128*64 bf16 = 147456 B

    prepack_weights<<<(9 * F_DIM * C_IN + 255) / 256, 256, 0, stream>>>(Wl, Wb);
    depthwise_kernel<<<(8 * C_IN * HW) / 256, 256, 0, stream>>>(x, out);
    learned_conv_wmma<<<dim3(256 / TW, 256, 8), 256, 0, stream>>>(x, Wb, bias, out);
}
